// StyleGAN2GeneratorBilinear_3968549781802
// MI455X (gfx1250) — matrix operand order repaired and checked by reading, not compiled, not run on hardware
//
#include <hip/hip_runtime.h>
#include <hip/hip_bf16.h>
#include <math.h>

// ---------------------------------------------------------------------------
// StyleGAN2 generator (bilinear) forward for gfx1250 / MI455X.
//
// All 3x3 modulated convs run as implicit GEMM on v_wmma_f32_16x16x32_f16
// (f16 in, fp32 accumulate). Inputs are style-modulated + (optionally)
// bilinear-2x-upsampled, stored f16 channels-last [b][pixel][cin]; weights
// pre-scaled f16 tap-major [cout][tap][cin]. With these layouts each lane's
// WMMA A/B fragments are exactly two global_load_b128 (CDNA5 16-bit A/B VGPR
// striping, ISA 7.12.2). The conv kernel uses 2x2 register blocking
// (32 cout x 32 px per wave, 4 accumulators) => 2 b128 loads per WMMA.
// Demod/noise/bias/fused-lrelu fused in the epilogue. Style MLP also on WMMA
// with LDS staging.
// ---------------------------------------------------------------------------

typedef __attribute__((ext_vector_type(16))) _Float16 v16h;
typedef __attribute__((ext_vector_type(8)))  _Float16 v8h;
typedef __attribute__((ext_vector_type(8)))  float    v8f;

#define CAT16(lo_, hi_) __builtin_shufflevector(lo_, hi_, 0,1,2,3,4,5,6,7,8,9,10,11,12,13,14,15)

#define SQRT2F 1.41421356237309515f

static __device__ __forceinline__ v8f wmma16(v16h A, v16h B, v8f C) {
  return __builtin_amdgcn_wmma_f32_16x16x32_f16(false, A, false, B, (short)0, C,
                                                false, false);
}

// ---------------------------------------------------------------------------
// Style MLP: rmsnorm(z) then 8x EqualLinear(512,512, lr_mul=0.01)+fused_lrelu.
// One block, 8 waves. x held in LDS (fp32 + f16 mirror). M=16 (batch 4 padded;
// rows independent so padding is harmless). Weights streamed from global and
// converted to f16 inline; accumulate fp32 via WMMA.
// ---------------------------------------------------------------------------
struct MlpArgs {
  const float* w[8];
  const float* b[8];
  const float* z;
  float* wout;   // [4][512] fp32
};

__global__ __launch_bounds__(256) void mlp_kernel(MlpArgs args) {
  __shared__ float    xs[16][512];
  __shared__ _Float16 xh[16][512];
  __shared__ float    rs[4];
  const int tid = threadIdx.x;

  // NormStyleCode: z * rsqrt(mean(z^2) + 1e-8)
  if (tid < 4) {
    const float* zr = args.z + tid * 512;
    float acc = 0.f;
    for (int i = 0; i < 512; ++i) acc += zr[i] * zr[i];
    rs[tid] = rsqrtf(acc * (1.0f / 512.0f) + 1e-8f);
  }
  __syncthreads();
  for (int i = tid; i < 16 * 512; i += 256) {
    int r = i >> 9, c = i & 511;
    xs[r][c] = (r < 4) ? args.z[r * 512 + c] * rs[r] : 0.f;
  }
  __syncthreads();

  const int lane = tid & 31;
  const int wave = tid >> 5;
  const int n  = lane & 15;
  const int hi = lane >> 4;
  const float wscale = 0.000441941738241592f;  // (1/sqrt(512)) * 0.01

  for (int layer = 0; layer < 8; ++layer) {
    // fp32 -> f16 mirror of current activations
    for (int i = tid; i < 16 * 512; i += 256) {
      int r = i >> 9, c = i & 511;
      xh[r][c] = (_Float16)xs[r][c];
    }
    __syncthreads();
    const float* W  = args.w[layer];
    const float* Bl = args.b[layer];
    for (int tI = 0; tI < 4; ++tI) {
      const int nb = (wave * 4 + tI) * 16;           // output-feature tile base
      v8f acc = {};
      const float* wr = W + (size_t)(nb + n) * 512;  // row = output feature
      for (int k0 = 0; k0 < 512; k0 += 32) {
        // A fragment (x, 16x32 f16): lane m=n, K = hi*8+0..7 and 16+hi*8+0..7
        v8h a0 = *(const v8h*)&xh[n][k0 + hi * 8];
        v8h a1 = *(const v8h*)&xh[n][k0 + 16 + hi * 8];
        v16h A = CAT16(a0, a1);
        // B fragment (w^T, 32x16 f16): lane col n, K = hi*8+0..7 and 16+hi*8+0..7
        const float* wp = wr + k0 + hi * 8;
        v16h Bm;
#pragma unroll
        for (int j = 0; j < 16; ++j) Bm[j] = (_Float16)(wp[j < 8 ? j : j + 8] * wscale);
        acc = wmma16(A, Bm, acc);
      }
      // D layout: lane holds col n, rows m = r + hi*8
      float bl = Bl[nb + n] * 0.01f;
#pragma unroll
      for (int r = 0; r < 8; ++r) {
        int m = r + hi * 8;
        float v = acc[r] + bl;
        v = SQRT2F * (v > 0.f ? v : 0.2f * v);
        xs[m][nb + n] = v;
      }
    }
    __syncthreads();
  }
  for (int i = tid; i < 4 * 512; i += 256)
    args.wout[i] = xs[i >> 9][i & 511];
}

// ---------------------------------------------------------------------------
// Per-layer style modulation: s[b][c] = <wstyle_b, mod_w_c> / sqrt(512) + mod_b
// ---------------------------------------------------------------------------
__global__ void style_kernel(const float* __restrict__ wstyle,
                             const float* __restrict__ mod_w,
                             const float* __restrict__ mod_b,
                             float* __restrict__ s, int Cin) {
  int c = blockIdx.x * 64 + threadIdx.x;
  int b = blockIdx.y;
  if (c >= Cin) return;
  const float* wr = mod_w + (size_t)c * 512;
  const float* zs = wstyle + b * 512;
  float acc = 0.f;
  for (int f = 0; f < 512; ++f) acc += zs[f] * wr[f];
  s[b * Cin + c] = acc * 0.0441941738241592f + mod_b[c];
}

// ---------------------------------------------------------------------------
// Weight prep: fp32 [cout][cin][3][3] -> f16 [cout][9][cin] (pre-scaled),
// plus wsum2[cout][cin] = sum_taps (w*scale)^2 for demodulation.
// ---------------------------------------------------------------------------
__global__ void wprep_kernel(const float* __restrict__ w, _Float16* __restrict__ wh,
                             float* __restrict__ wsum2, int Cin, float scale) {
  int ci = blockIdx.x * 64 + threadIdx.x;
  int co = blockIdx.y;
  if (ci >= Cin) return;
  const float* wp = w + ((size_t)co * Cin + ci) * 9;
  float s2 = 0.f;
#pragma unroll
  for (int t = 0; t < 9; ++t) {
    float v = wp[t] * scale;
    s2 += v * v;
    wh[((size_t)co * 9 + t) * Cin + ci] = (_Float16)v;
  }
  wsum2[(size_t)co * Cin + ci] = s2;
}

// demod[b][co] = rsqrt(sum_ci s^2 * wsum2 + 1e-8)
__global__ void demod_kernel(const float* __restrict__ s, const float* __restrict__ wsum2,
                             float* __restrict__ demod, int Cin, int Cout) {
  int co = blockIdx.x * 64 + threadIdx.x;
  int b = blockIdx.y;
  if (co >= Cout) return;
  const float* sr = s + b * Cin;
  const float* wr = wsum2 + (size_t)co * Cin;
  float acc = 0.f;
  for (int ci = 0; ci < Cin; ++ci) { float sv = sr[ci]; acc += sv * sv * wr[ci]; }
  demod[b * Cout + co] = rsqrtf(acc + 1e-8f);
}

// ---------------------------------------------------------------------------
// Input prep: (optional bilinear 2x upsample) * style, fp32 -> f16,
// written channels-last [b][pixel][cin] via LDS transpose (coalesced both ways).
// Bilinear: src = out*0.5 - 0.25 (half-pixel centers), indices clamped.
// ---------------------------------------------------------------------------
__global__ __launch_bounds__(256)
void prep_kernel(const float* __restrict__ in, const float* __restrict__ s,
                 _Float16* __restrict__ out, int Cin, int inH, int inW,
                 int upsample, unsigned long long in_bstride) {
  __shared__ _Float16 tile[32][33];
  const int b = blockIdx.z;
  const int ciBase = blockIdx.y * 32;
  const int outH = upsample ? inH * 2 : inH;
  const int outW = upsample ? inW * 2 : inW;
  const int P = outH * outW;
  const int inP = inH * inW;
  const int pxBase = blockIdx.x * 32;
  const int px  = threadIdx.x & 31;
  const int ci8 = threadIdx.x >> 5;  // 0..7
  const int pix = pxBase + px;
  const float* ip = in + (size_t)in_bstride * b;

  for (int cc = 0; cc < 4; ++cc) {
    int ci = ciBase + cc * 8 + ci8;
    float val = 0.f;
    if (pix < P) {
      if (!upsample) {
        val = ip[(size_t)ci * inP + pix];
      } else {
        int oy = pix / outW, ox = pix % outW;
        float fy = oy * 0.5f - 0.25f, fx = ox * 0.5f - 0.25f;
        int y0 = (int)floorf(fy); float wy = fy - (float)y0;
        int x0 = (int)floorf(fx); float wx = fx - (float)x0;
        int y0c = y0 < 0 ? 0 : (y0 > inH - 1 ? inH - 1 : y0);
        int y1t = y0 + 1; int y1c = y1t < 0 ? 0 : (y1t > inH - 1 ? inH - 1 : y1t);
        int x0c = x0 < 0 ? 0 : (x0 > inW - 1 ? inW - 1 : x0);
        int x1t = x0 + 1; int x1c = x1t < 0 ? 0 : (x1t > inW - 1 ? inW - 1 : x1t);
        const float* base = ip + (size_t)ci * inP;
        float v00 = base[y0c * inW + x0c], v01 = base[y0c * inW + x1c];
        float v10 = base[y1c * inW + x0c], v11 = base[y1c * inW + x1c];
        val = (1.f - wy) * ((1.f - wx) * v00 + wx * v01) +
              wy * ((1.f - wx) * v10 + wx * v11);
      }
      val *= s[b * Cin + ci];
    }
    tile[cc * 8 + ci8][px] = (_Float16)val;
  }
  __syncthreads();
  const int ci2  = threadIdx.x & 31;   // fastest -> contiguous cin on write
  const int pxl0 = threadIdx.x >> 5;
  for (int k = 0; k < 4; ++k) {
    int pxl = pxl0 + k * 8;
    int pixw = pxBase + pxl;
    if (pixw < P)
      out[((size_t)b * P + pixw) * Cin + ciBase + ci2] = tile[ci2][pxl];
  }
}

// ---------------------------------------------------------------------------
// Modulated 3x3 conv as implicit GEMM on WMMA f16->f32, 2x2 register blocked.
// Block = 128 threads = 4 waves; each wave computes a 32(cout) x 32(pixel)
// macro-tile with 4 independent accumulators. Per K-step (tap, cin32):
// 2 A fragments + 2 B fragments (8 x b128) feed 4 WMMAs => 2 loads/WMMA.
// Zero-padding / partial-P handled by predicated B loads + gated stores;
// WMMA always executes with full EXEC. Epilogue fuses demod, +noise, +bias,
// sqrt(2)*leaky_relu(0.2); fp32 out [b][co][p].
// ---------------------------------------------------------------------------
__global__ __launch_bounds__(128)
void conv_kernel(const _Float16* __restrict__ xf16, const _Float16* __restrict__ wf16,
                 const float* __restrict__ demod, const float* __restrict__ noise,
                 const float* __restrict__ noise_strength, const float* __restrict__ act_b,
                 float* __restrict__ out, int Cin, int Cout, int H, int W, int wshift) {
  const int P = H * W;
  const int lane = threadIdx.x & 31;
  const int wave = threadIdx.x >> 5;
  const int n  = lane & 15;
  const int hi = lane >> 4;
  const int b = blockIdx.z;
  const int coutBase = blockIdx.y << 5;              // 32 couts per wave
  const int pixBase  = blockIdx.x * 128 + wave * 32; // 32 pixels per wave
  if (pixBase >= P) return;                          // wave-uniform; no barriers
  const int  pix0 = pixBase + n;                     // P is a multiple of 16
  const int  pix1 = pixBase + 16 + n;
  const bool p1ok = (pixBase + 16) < P;              // wave-uniform
  const int y0 = pix0 >> wshift, x0c_ = pix0 & (W - 1);
  const int y1 = pix1 >> wshift, x1c_ = pix1 & (W - 1);

  v8f acc00 = {}, acc01 = {}, acc10 = {}, acc11 = {};
  // A rows for this lane (m = n): weights [cout][tap][cin], +hi*16 K offset
  const _Float16* wrow0 = wf16 + (size_t)(coutBase + n) * 9 * Cin + hi * 16;
  const _Float16* wrow1 = wf16 + (size_t)(coutBase + 16 + n) * 9 * Cin + hi * 16;
  const size_t xrowB = (size_t)b * P * (size_t)Cin;

#pragma unroll
  for (int t = 0; t < 9; ++t) {
    const int ky = t / 3 - 1;
    const int kx = t % 3 - 1;
    const int sy0 = y0 + ky, sx0 = x0c_ + kx;
    const int sy1 = y1 + ky, sx1 = x1c_ + kx;
    const bool v0 = ((unsigned)sy0 < (unsigned)H) && ((unsigned)sx0 < (unsigned)W);
    const bool v1 = p1ok && ((unsigned)sy1 < (unsigned)H) && ((unsigned)sx1 < (unsigned)W);
    const int sp0 = v0 ? (sy0 * W + sx0) : 0;
    const int sp1 = v1 ? (sy1 * W + sx1) : 0;
    const _Float16* xr0 = xf16 + xrowB + (size_t)sp0 * Cin + (hi << 4);
    const _Float16* xr1 = xf16 + xrowB + (size_t)sp1 * Cin + (hi << 4);
    const _Float16* wr0 = wrow0 + (size_t)t * Cin;
    const _Float16* wr1 = wrow1 + (size_t)t * Cin;
    // warm L2/L0 for this tap's streamed input rows (global_prefetch_b8)
    if (v0) __builtin_prefetch((const void*)xr0, 0, 1);
    if (v1) __builtin_prefetch((const void*)xr1, 0, 1);
    for (int c0 = 0; c0 < Cin; c0 += 32) {
      // A: K = c0 + hi*16 + 0..15
      v16h A0 = CAT16(*(const v8h*)(wr0 + c0), *(const v8h*)(wr0 + c0 + 8));
      v16h A1 = CAT16(*(const v8h*)(wr1 + c0), *(const v8h*)(wr1 + c0 + 8));
      // B: K = c0 + hi*16 + 0..15 (contiguous channels-last)
      v8h b00 = {}, b01 = {}, b10 = {}, b11 = {};
      if (v0) { b00 = *(const v8h*)(xr0 + c0); b01 = *(const v8h*)(xr0 + c0 + 8); }
      if (v1) { b10 = *(const v8h*)(xr1 + c0); b11 = *(const v8h*)(xr1 + c0 + 8); }
      v16h B0 = CAT16(b00, b01);
      v16h B1 = CAT16(b10, b11);
      acc00 = wmma16(A0, B0, acc00);
      acc01 = wmma16(A0, B1, acc01);
      acc10 = wmma16(A1, B0, acc10);
      acc11 = wmma16(A1, B1, acc11);
    }
  }

  const float ns = noise_strength[0];
  const float nz0 = noise[(size_t)b * P + pix0];
  const float nz1 = p1ok ? noise[(size_t)b * P + pix1] : 0.f;

  auto epi = [&](const v8f& acc, int coutOff, int pix, float nz, bool ok) {
    if (!ok) return;
#pragma unroll
    for (int r = 0; r < 8; ++r) {
      int co = coutBase + coutOff + r + hi * 8;
      float v = acc[r] * demod[b * Cout + co] + ns * nz + act_b[co];
      v = SQRT2F * (v > 0.f ? v : 0.2f * v);
      out[((size_t)b * Cout + co) * P + pix] = v;
    }
  };
  epi(acc00, 0,  pix0, nz0, true);
  epi(acc01, 0,  pix1, nz1, p1ok);
  epi(acc10, 16, pix0, nz0, true);
  epi(acc11, 16, pix1, nz1, p1ok);
}

// ---------------------------------------------------------------------------
// to_rgb: 1x1 modulated conv (no demod, M=3 so plain VALU), + bias,
// + fused bilinear-2x upsample of the previous skip. Coalesced over pixels.
// ---------------------------------------------------------------------------
__global__ __launch_bounds__(256)
void torgb_kernel(const float* __restrict__ xin, const float* __restrict__ s,
                  const float* __restrict__ wrgb, const float* __restrict__ bias,
                  const float* __restrict__ skip, float* __restrict__ outp,
                  int Cin, int H, int W, int hasSkip) {
  const int P = H * W;
  const int pix = blockIdx.x * 256 + threadIdx.x;
  const int b = blockIdx.y;
  if (pix >= P) return;
  float a0 = 0.f, a1 = 0.f, a2 = 0.f;
  const float* xb = xin + (size_t)b * Cin * P + pix;
  const float* sb = s + b * Cin;
  for (int ci = 0; ci < Cin; ++ci) {
    float m = xb[(size_t)ci * P] * sb[ci];
    a0 += m * wrgb[ci];
    a1 += m * wrgb[Cin + ci];
    a2 += m * wrgb[2 * Cin + ci];
  }
  const float scale = rsqrtf((float)Cin);
  float o[3] = { a0 * scale + bias[0], a1 * scale + bias[1], a2 * scale + bias[2] };
  if (hasSkip) {
    int inH = H >> 1, inW = W >> 1, inP = inH * inW;
    int oy = pix / W, ox = pix % W;
    float fy = oy * 0.5f - 0.25f, fx = ox * 0.5f - 0.25f;
    int y0 = (int)floorf(fy); float wy = fy - (float)y0;
    int x0 = (int)floorf(fx); float wx = fx - (float)x0;
    int y0c = y0 < 0 ? 0 : (y0 > inH - 1 ? inH - 1 : y0);
    int y1t = y0 + 1; int y1c = y1t > inH - 1 ? inH - 1 : (y1t < 0 ? 0 : y1t);
    int x0c = x0 < 0 ? 0 : (x0 > inW - 1 ? inW - 1 : x0);
    int x1t = x0 + 1; int x1c = x1t > inW - 1 ? inW - 1 : (x1t < 0 ? 0 : x1t);
#pragma unroll
    for (int c = 0; c < 3; ++c) {
      const float* sp = skip + ((size_t)b * 3 + c) * inP;
      float v00 = sp[y0c * inW + x0c], v01 = sp[y0c * inW + x1c];
      float v10 = sp[y1c * inW + x0c], v11 = sp[y1c * inW + x1c];
      o[c] += (1.f - wy) * ((1.f - wx) * v00 + wx * v01) +
              wy * ((1.f - wx) * v10 + wx * v11);
    }
  }
#pragma unroll
  for (int c = 0; c < 3; ++c)
    outp[((size_t)b * 3 + c) * P + pix] = o[c];
}

// ---------------------------------------------------------------------------
// Host orchestration.
//
// Input flattening assumption (recursive insertion order of setup_inputs()):
//  [0]=z, [1..13]=noise0..noise12, then params:
//  [14..21]=mlp_w[0..7], [22..29]=mlp_b[0..7], [30]=const,
//  [31..35]=conv1{mod_w,mod_b,w,noise_strength,act_b},
//  [36..39]=to_rgb1{mod_w,mod_b,w,bias},
//  [40..99]=convs[12]x{mod_w,mod_b,w,noise_strength,act_b},
//  [100..123]=to_rgbs[6]x{mod_w,mod_b,w,bias}.
//
// Workspace layout (needs ~430 MB).
// ---------------------------------------------------------------------------
extern "C" void kernel_launch(void* const* d_in, const int* in_sizes, int n_in,
                              void* d_out, int out_size, void* d_ws, size_t ws_size,
                              hipStream_t stream) {
  (void)in_sizes; (void)n_in; (void)out_size; (void)ws_size;
  auto F = [&](int i) { return (const float*)d_in[i]; };

  const float* z = F(0);
  const float* noise[13];
  for (int i = 0; i < 13; ++i) noise[i] = F(1 + i);

  struct ConvP { const float *mod_w, *mod_b, *w, *ns, *act_b; };
  struct RgbP  { const float *mod_w, *mod_b, *w, *bias; };

  MlpArgs ma;
  for (int i = 0; i < 8; ++i) { ma.w[i] = F(14 + i); ma.b[i] = F(22 + i); }
  ma.z = z;
  const float* constp = F(30);
  ConvP conv1 = { F(31), F(32), F(33), F(34), F(35) };
  RgbP  rgb1  = { F(36), F(37), F(38), F(39) };
  ConvP convs[12];
  for (int i = 0; i < 12; ++i) {
    int b = 40 + 5 * i;
    convs[i] = { F(b), F(b + 1), F(b + 2), F(b + 3), F(b + 4) };
  }
  RgbP rgbs[6];
  for (int i = 0; i < 6; ++i) {
    int b = 100 + 4 * i;
    rgbs[i] = { F(b), F(b + 1), F(b + 2), F(b + 3) };
  }

  // workspace carve-up
  char* ws = (char*)d_ws;
  const size_t MB = 1024ull * 1024ull;
  float*    wstyle = (float*)(ws + 0);            // 8 KB  [4][512]
  float*    sconv  = (float*)(ws + 64 * 1024);    // 8 KB  [4][Cin]
  float*    srgb   = (float*)(ws + 128 * 1024);   // 8 KB
  float*    demodb = (float*)(ws + 192 * 1024);   // 8 KB  [4][Cout]
  float*    wsum2  = (float*)(ws + 1 * MB);       // <=1 MB [Cout][Cin]
  _Float16* wf16   = (_Float16*)(ws + 4 * MB);    // <=4.6 MB [Cout][9][Cin]
  float*    skipA  = (float*)(ws + 16 * MB);      // <=3 MB
  float*    skipB  = (float*)(ws + 20 * MB);      // <=3 MB
  _Float16* xf16   = (_Float16*)(ws + 32 * MB);   // <=128 MB [b][P][Cin]
  float*    actA   = (float*)(ws + 168 * MB);     // <=128 MB
  float*    actB   = (float*)(ws + 300 * MB);     // <=128 MB

  ma.wout = wstyle;
  mlp_kernel<<<1, 256, 0, stream>>>(ma);

  auto run_conv = [&](const ConvP& cp, const float* xin, unsigned long long bstride,
                      int Cin, int inH, int Cout, const float* nz, bool up, float* outp) {
    int outH = up ? inH * 2 : inH;
    int W = outH, P = outH * outH;
    int wshift = 0; while ((1 << wshift) < W) ++wshift;
    float wsc = 1.0f / sqrtf((float)(Cin * 9));
    wprep_kernel<<<dim3((Cin + 63) / 64, Cout), 64, 0, stream>>>(cp.w, wf16, wsum2, Cin, wsc);
    style_kernel<<<dim3((Cin + 63) / 64, 4), 64, 0, stream>>>(wstyle, cp.mod_w, cp.mod_b, sconv, Cin);
    demod_kernel<<<dim3((Cout + 63) / 64, 4), 64, 0, stream>>>(sconv, wsum2, demodb, Cin, Cout);
    prep_kernel<<<dim3((P + 31) / 32, Cin / 32, 4), 256, 0, stream>>>(
        xin, sconv, xf16, Cin, inH, inH, up ? 1 : 0, bstride);
    conv_kernel<<<dim3((P + 127) / 128, Cout / 32, 4), 128, 0, stream>>>(
        xf16, wf16, demodb, nz, cp.ns, cp.act_b, outp, Cin, Cout, outH, W, wshift);
  };

  auto run_rgb = [&](const RgbP& rp, const float* xin, int Cin, int H,
                     const float* skipin, float* outp) {
    style_kernel<<<dim3((Cin + 63) / 64, 4), 64, 0, stream>>>(wstyle, rp.mod_w, rp.mod_b, srgb, Cin);
    torgb_kernel<<<dim3((H * H + 255) / 256, 4), 256, 0, stream>>>(
        xin, srgb, rp.w, rp.bias, skipin, outp, Cin, H, H, skipin ? 1 : 0);
  };

  // conv1 @ 4x4 (input = const, batch-broadcast => bstride 0)
  run_conv(conv1, constp, 0ull, 512, 4, 512, noise[0], false, actA);
  run_rgb(rgb1, actA, 512, 4, nullptr, skipA);

  float* cur = actA;           // holds current conv output (input to next stage)
  float* other = actB;
  float* skipCur = skipA;
  float* skipOther = skipB;
  int res = 4, cin = 512;

  for (int j = 0; j < 6; ++j) {
    int r = 8 << j;
    int cout = (r <= 64) ? 512 : (r == 128 ? 256 : 128);
    // upsample conv: res -> r
    run_conv(convs[2 * j], cur, (unsigned long long)cin * res * res,
             cin, res, cout, noise[1 + 2 * j], true, other);
    // same-res conv at r
    run_conv(convs[2 * j + 1], other, (unsigned long long)cout * r * r,
             cout, r, cout, noise[2 + 2 * j], false, cur);
    // to_rgb with fused skip upsample; last stage writes d_out directly
    float* rgbOut = (j == 5) ? (float*)d_out : skipOther;
    run_rgb(rgbs[j], cur, cout, r, skipCur, rgbOut);
    float* t = skipCur; skipCur = skipOther; skipOther = t;
    cin = cout;
    res = r;
  }
}
